// Net_74629351735532
// MI455X (gfx1250) — matrix operand order repaired and checked by reading, not compiled, not run on hardware
//
#include <hip/hip_runtime.h>
#include <hip/hip_bf16.h>
#include <math.h>

typedef __attribute__((ext_vector_type(16))) __bf16 v16bf;
typedef __attribute__((ext_vector_type(8)))  float  v8f;
typedef __attribute__((ext_vector_type(4)))  unsigned int u32x4;
typedef __attribute__((ext_vector_type(8)))  int  i32x8;
typedef __attribute__((ext_vector_type(4)))  int  i32x4;

#if defined(__HIP_DEVICE_COMPILE__) && __has_builtin(__builtin_amdgcn_tensor_load_to_lds)
#define HAVE_TDM 1
#else
#define HAVE_TDM 0
#endif

constexpr int NN = 50000;   // nodes
constexpr int NE = 800000;  // edges
constexpr int D  = 64;      // feature dim (both layers)
constexpr int NC = 8;       // clusters
constexpr int NT = NN / 16; // 3125 row tiles (exact)

// ---------------------------------------------------------------- degree
__global__ void k_init_deg(float* __restrict__ deg, int n) {
    int i = blockIdx.x * 256 + threadIdx.x;
    if (i < n) deg[i] = 1.0f;                     // self-loop weight
}

__global__ void k_deg_accum(const long long* __restrict__ dst,
                            const float* __restrict__ w,
                            float* __restrict__ deg, int e) {
    int i = blockIdx.x * 256 + threadIdx.x;
    if (i < e) atomicAdd(&deg[(int)dst[i]], w[i]);
}

__global__ void k_dinv(float* __restrict__ deg, int n) {
    int i = blockIdx.x * 256 + threadIdx.x;
    if (i < n) {
        float d = deg[i];
        deg[i] = (d > 0.0f) ? rsqrtf(d) : 0.0f;   // in-place -> dinv
    }
}

// ---------------- WMMA GEMM + fused epilogue:
//   H = [relu](X) @ W          (needed by the edge scatter)
//   A = bias + dinv^2 * H      (self-loop + bias init of the accumulator)
// One wave = one 16-row tile: 4 column tiles x 2 K-steps = 8 WMMAs.
// A-tile staged into LDS as f32 by the Tensor Data Mover (2D D#, 4KB tile),
// overlapped with cooperative bf16 staging of W^T; bf16 conversion happens
// during fragment gather (v_cvt_pk_bf16_f32).
template <bool RELU>
__global__ __launch_bounds__(256) void k_gemm64_bf16(
        const float* __restrict__ X, const float* __restrict__ W,
        const float* __restrict__ bias, const float* __restrict__ dinv,
        float* __restrict__ H, float* __restrict__ A, int ntiles) {
    __shared__ __bf16 sWt[D * D];        // W transposed bf16: sWt[n*64 + k]
    __shared__ float  sAf[8][16 * D];    // per-wave f32 A tile, row-major [m][k]

    const int tid  = threadIdx.x;
    const int wave = tid >> 5;
    const int lane = tid & 31;
    const int tile = blockIdx.x * 8 + wave;

#if HAVE_TDM
    if (tile < ntiles) {
        // Tensor DMA: contiguous 16x64 f32 tile -> this wave's LDS region.
        unsigned long long ga =
            (unsigned long long)(const void*)(X + (size_t)tile * 16 * D);
        unsigned lds = (unsigned)(size_t)(&sAf[wave][0]);
        u32x4 g0;
        g0[0] = 1u;                                          // count=1 (valid)
        g0[1] = lds;                                         // lds_addr
        g0[2] = (unsigned)(ga & 0xFFFFFFFFull);              // global_addr lo
        g0[3] = (unsigned)((ga >> 32) & 0x01FFFFFFull) | (2u << 30); // hi|type=2
        i32x8 g1;
        g1[0] = (int)(2u << 16);     // data_size=2 -> 4 bytes; wg_mask=0
        g1[1] = (int)(64u << 16);    // tensor_dim0 = 64 (bits 79:48)
        g1[2] = (int)(16u << 16);    // tensor_dim1 = 16 (bits 111:80)
        g1[3] = (int)(64u << 16);    // tile_dim0   = 64 (bits 127:112)
        g1[4] = 16;                  // tile_dim1   = 16 (bits 143:128)
        g1[5] = 64;                  // tensor_dim0_stride = 64 (bits 207:160)
        g1[6] = 0;
        g1[7] = 0;
        i32x4 gz4 = {0, 0, 0, 0};              // 2D tensor: groups 2/3 unused
        i32x8 gz8 = {0, 0, 0, 0, 0, 0, 0, 0};  // extra operand (clang-23 form)
        __builtin_amdgcn_tensor_load_to_lds(g0, g1, gz4, gz4, gz8, 0);
    }
#else
    if (tile < ntiles) {
        const float4* src = (const float4*)(X + (size_t)tile * 16 * D);
        float4* dst4 = (float4*)&sAf[wave][0];
        for (int t = lane; t < 16 * D / 4; t += 32) dst4[t] = src[t];
    }
#endif

    // Stage W transposed as bf16 (whole block cooperates; overlaps the DMA).
    for (int t = tid; t < D * D; t += 256) {
        int k = t >> 6, n = t & 63;
        sWt[n * D + k] = (__bf16)W[t];
    }

#if HAVE_TDM
    __builtin_amdgcn_s_wait_tensorcnt(0);    // own tile resident in LDS
#endif
    __syncthreads();                         // sWt visible to all waves
    if (tile >= ntiles) return;              // uniform per wave; EXEC all-1s

    const int M    = lane & 15;              // row in tile / col in ntile
    const int half = lane >> 4;
    const float* arow = &sAf[wave][M * D];

    v8f acc[4] = {v8f{0}, v8f{0}, v8f{0}, v8f{0}};

    #pragma unroll
    for (int kb = 0; kb < 2; ++kb) {
        // A fragment, ISA 16-bit A layout: VGPR v holds K pair
        //   K = kb*32 + (v/4)*16 + half*8 + (v%4)*2
        v16bf a;
        #pragma unroll
        for (int v = 0; v < 8; ++v) {
            int K = kb * 32 + ((v >> 2) << 4) + (half << 3) + ((v & 3) << 1);
            float f0 = arow[K], f1 = arow[K + 1];
            if (RELU) { f0 = fmaxf(f0, 0.0f); f1 = fmaxf(f1, 0.0f); }
            a[2 * v]     = (__bf16)f0;
            a[2 * v + 1] = (__bf16)f1;
        }
        #pragma unroll
        for (int n = 0; n < 4; ++n) {
            // B fragment (32x16): same K order as A: K=(v/4)*16+half*8+(v%4)*2.
            v16bf b;
            const int col = n * 16 + M;
            #pragma unroll
            for (int v = 0; v < 8; ++v) {
                int K = kb * 32 + ((v >> 2) << 4) + (half << 3) + ((v & 3) << 1);
                b[2 * v]     = sWt[col * D + K];
                b[2 * v + 1] = sWt[col * D + K + 1];
            }
            acc[n] = __builtin_amdgcn_wmma_f32_16x16x32_bf16(
                false, a, false, b, (short)0, acc[n], false, false);
        }
    }

    // Fused epilogue. C/D layout: VGPR r, lane -> row (r+8*half), col (n*16+M).
    float bn[4];
    #pragma unroll
    for (int n = 0; n < 4; ++n) bn[n] = bias[n * 16 + M];
    float d2[8];
    #pragma unroll
    for (int r = 0; r < 8; ++r) {
        float di = dinv[tile * 16 + (half << 3) + r];
        d2[r] = di * di;
    }
    float* hdst = H + (size_t)tile * 16 * D;
    float* adst = A + (size_t)tile * 16 * D;
    #pragma unroll
    for (int n = 0; n < 4; ++n)
        #pragma unroll
        for (int r = 0; r < 8; ++r) {
            float v = acc[n][r];
            int idx = ((half << 3) + r) * D + n * 16 + M;
            hdst[idx] = v;
            adst[idx] = bn[n] + d2[r] * v;
        }
}

// ------------------- edge scatter: out[dst] += dinv[s]*w*dinv[d] * h[src]
// 4 edges per wave, 2 features per lane; indices loaded up front and source
// rows prefetched (global_prefetch) to hide the random-gather latency.
__global__ __launch_bounds__(256) void k_scatter(
        const long long* __restrict__ srcI, const long long* __restrict__ dstI,
        const float* __restrict__ w, const float* __restrict__ dinv,
        const float* __restrict__ h, float* __restrict__ out, int e) {
    constexpr int EPW = 4;
    long long gt = (long long)blockIdx.x * 256 + threadIdx.x;
    int wv   = (int)(gt >> 5);
    int lane = (int)(gt & 31);
    int e0 = wv * EPW;
    if (e0 >= e) return;
    int cnt = (e - e0 < EPW) ? (e - e0) : EPW;   // uniform per wave

    int se[EPW], de[EPW];
    float nw[EPW];
    for (int j = 0; j < cnt; ++j) {
        se[j] = (int)srcI[e0 + j];
        de[j] = (int)dstI[e0 + j];
        nw[j] = w[e0 + j];
        __builtin_prefetch(h + (size_t)se[j] * D + lane * 2, 0, 0);
    }
    for (int j = 0; j < cnt; ++j) {
        float norm = dinv[se[j]] * nw[j] * dinv[de[j]];
        float2 v = ((const float2*)(h + (size_t)se[j] * D))[lane];
        float* o = out + (size_t)de[j] * D + lane * 2;
        atomicAdd(o + 0, v.x * norm);
        atomicAdd(o + 1, v.y * norm);
    }
}

// ------------------- s = relu(a2) @ Wm + bm ; out = softmax(s) per row (8-wide)
__global__ __launch_bounds__(256) void k_final(const float* __restrict__ a2,
                                               const float* __restrict__ Wm,
                                               const float* __restrict__ bm,
                                               float* __restrict__ out, int n) {
    __shared__ float sW[D * NC];
    __shared__ float sb[NC];
    int tid = threadIdx.x;
    for (int t = tid; t < D * NC; t += 256) sW[t] = Wm[t];
    if (tid < NC) sb[tid] = bm[tid];
    __syncthreads();

    int r = blockIdx.x * 256 + tid;
    if (r >= n) return;
    const float* row = a2 + (size_t)r * D;

    float s[NC];
    #pragma unroll
    for (int c = 0; c < NC; ++c) s[c] = sb[c];
    for (int k = 0; k < D; ++k) {
        float z = fmaxf(row[k], 0.0f);
        #pragma unroll
        for (int c = 0; c < NC; ++c) s[c] = fmaf(z, sW[k * NC + c], s[c]);
    }
    float m = s[0];
    #pragma unroll
    for (int c = 1; c < NC; ++c) m = fmaxf(m, s[c]);
    float sum = 0.0f;
    #pragma unroll
    for (int c = 0; c < NC; ++c) { s[c] = __expf(s[c] - m); sum += s[c]; }
    float inv = 1.0f / sum;
    float* o = out + (size_t)r * NC;
    #pragma unroll
    for (int c = 0; c < NC; ++c) o[c] = s[c] * inv;
}

// ---------------------------------------------------------------- launcher
extern "C" void kernel_launch(void* const* d_in, const int* in_sizes, int n_in,
                              void* d_out, int out_size, void* d_ws, size_t ws_size,
                              hipStream_t stream) {
    const float*     x   = (const float*)d_in[0];
    const long long* ei  = (const long long*)d_in[1];   // [2, E] int64
    const float*     ew  = (const float*)d_in[2];
    const float*     W1  = (const float*)d_in[3];
    const float*     b1  = (const float*)d_in[4];
    const float*     W2  = (const float*)d_in[5];
    const float*     b2  = (const float*)d_in[6];
    const float*     Wm  = (const float*)d_in[7];
    const float*     bm  = (const float*)d_in[8];
    float* out = (float*)d_out;

    float* dinv = (float*)d_ws;            // N
    float* h    = dinv + NN;               // N*64
    float* a    = h + (size_t)NN * D;      // N*64

    const long long* srcI = ei;            // edge_index[0]
    const long long* dstI = ei + NE;       // edge_index[1]

    const int blkN   = (NN + 255) / 256;
    const int blkE   = (NE + 255) / 256;
    const int blkSc  = (int)((((long long)(NE + 3) / 4) * 32 + 255) / 256);
    const int blkGem = (NT + 7) / 8;

    // gcn_norm
    k_init_deg <<<blkN,  256, 0, stream>>>(dinv, NN);
    k_deg_accum<<<blkE,  256, 0, stream>>>(dstI, ew, dinv, NE);
    k_dinv     <<<blkN,  256, 0, stream>>>(dinv, NN);

    // layer 1: h1 = x @ W1 ; a1 = b1 + dinv^2*h1 (fused) + scatter(norm*h1)
    k_gemm64_bf16<false><<<blkGem, 256, 0, stream>>>(x, W1, b1, dinv, h, a, NT);
    k_scatter  <<<blkSc, 256, 0, stream>>>(srcI, dstI, ew, dinv, h, a, NE);

    // layer 2: h2 = relu(a1) @ W2 ; a2 = b2 + dinv^2*h2 (fused) + scatter
    k_gemm64_bf16<true> <<<blkGem, 256, 0, stream>>>(a, W2, b2, dinv, h, a, NT);
    k_scatter  <<<blkSc, 256, 0, stream>>>(srcI, dstI, ew, dinv, h, a, NE);

    // projection + softmax
    k_final    <<<blkN,  256, 0, stream>>>(a, Wm, bm, out, NN);
}
